// snn_1_46729244181192
// MI455X (gfx1250) — matrix operand order repaired and checked by reading, not compiled, not run on hardware
//
#include <hip/hip_runtime.h>
#include <hip/hip_bf16.h>

// ---------------------------------------------------------------------------
// SNN forward (pool -> fc_in GEMM -> T x [LIF, hid GEMM+LIF, out GEMM+LIF])
// MI455X / gfx1250: wave32, v_wmma_f32_16x16x32_f16 for the two large GEMMs.
// Memory-bound problem (182MB input stream, ~4.3 GFLOP total), so the layout
// goal is: one streaming pass over x, one batched WMMA GEMM for all T, and
// LIF fused into GEMM epilogues for the per-step chain.
// ---------------------------------------------------------------------------

typedef __attribute__((ext_vector_type(16))) _Float16 v16h;
typedef __attribute__((ext_vector_type(8)))  _Float16 v8h;
typedef __attribute__((ext_vector_type(8)))  float    v8f;

union V16U { v16h v; v8h h[2]; };

#define T_STEPS 8
#define BATCH   64
#define CCH     3
#define DPOOL   14
#define NSEG    42                 // CCH * DPOOL
#define H0      256
#define H1      128
#define NCLS    6
#define KIN     256                // input_dim (Hp*Wp = 16*16)
#define MFULL   (T_STEPS*BATCH*NSEG)   // 21504 rows for fc_in
#define MSTEP   (BATCH*NSEG)           // 2688 rows per timestep
#define FLATK   (NSEG*H1)              // 5376

__device__ __forceinline__ float clamp01f(float b) {
    return fminf(fmaxf(b, 0.f), 1.f);
}

// ---------------------------------------------------------------------------
// generic helpers
// ---------------------------------------------------------------------------
__global__ void __launch_bounds__(256)
k_f32_to_f16(const float* __restrict__ src, _Float16* __restrict__ dst, int n) {
    int i = blockIdx.x * 256 + threadIdx.x;
    if (i < n) dst[i] = (_Float16)src[i];
}

__global__ void __launch_bounds__(256)
k_zero_f32(float* __restrict__ p, int n) {
    int i = blockIdx.x * 256 + threadIdx.x;
    if (i < n) p[i] = 0.f;
}

// ---------------------------------------------------------------------------
// 2x2x2 avg pool, VALID, writing f16 A-operand pr[(t*B+b)*42+n][256]
// x: (B, C, 232, 32, 32) f32
// ---------------------------------------------------------------------------
__global__ void __launch_bounds__(256)
k_pool(const float* __restrict__ x, _Float16* __restrict__ pr, int total) {
    int idx = blockIdx.x * 256 + threadIdx.x;
    if (idx >= total) return;
    int f  = idx & 255;           // hp*16 + wp
    int r  = idx >> 8;            // (t*64+b)*42 + n
    int n  = r % NSEG;
    int tb = r / NSEG;
    int b  = tb & 63;
    int t  = tb >> 6;
    int c  = n / DPOOL;
    int d  = n % DPOOL;
    int hp = f >> 4, wp = f & 15;
    int dd = t * 29 + 2 * d;      // chunk = 29, VALID drops last slice
    const float* base = x + ((((size_t)(b * CCH + c) * 232 + dd) * 32 + 2 * hp) * 32 + 2 * wp);
    float s = 0.f;
#pragma unroll
    for (int dz = 0; dz < 2; ++dz)
#pragma unroll
        for (int dy = 0; dy < 2; ++dy) {
            const float* p = base + dz * 1024 + dy * 32;
            s += p[0] + p[1];
        }
    pr[idx] = (_Float16)(s * 0.125f);
}

// ---------------------------------------------------------------------------
// WMMA tile fragment loaders, wave32, contiguous k order chosen by this code for both A and B.
// A (16x32 f16): lane=(kgrp<<4)|m, elem e -> K = 16*kgrp+e
//   => two contiguous 16B chunks per lane per K-step.
// B (32x16 f16): lane=(kgrp<<4)|n, elem e -> K = 16*kgrp+e
//   => one contiguous 32B chunk per lane (W stored row-major over K).
// ---------------------------------------------------------------------------
__device__ __forceinline__ v16h load_fragA(const _Float16* __restrict__ arow, int kk) {
    V16U a;
    a.h[0] = *(const v8h*)(arow + kk);        // K = kk + 16*kgrp + 0..7
    a.h[1] = *(const v8h*)(arow + kk + 8);    // K = kk + 16*kgrp + 8..15
    return a.v;
}

// ---------------------------------------------------------------------------
// fc_in: C[M,N] = A[M,K] * W[N,K]^T + bias[N], f16 in / f32 out
// One 16x16 tile per wave, 8 waves per block, exact tiling (no guards -> EXEC
// all-ones at every WMMA).
// ---------------------------------------------------------------------------
__global__ void __launch_bounds__(256)
k_gemm_fcin(const _Float16* __restrict__ A, const _Float16* __restrict__ Bw,
            const float* __restrict__ bias, float* __restrict__ C,
            int M, int N, int K) {
    const int lane = threadIdx.x & 31;
    const int wave = threadIdx.x >> 5;
    const int tile = blockIdx.x * 8 + wave;
    const int ntn  = N >> 4;
    const int row0 = (tile / ntn) << 4;
    const int col0 = (tile % ntn) << 4;
    const int kgrp = lane >> 4;
    const int l15  = lane & 15;

    const _Float16* arow = A  + (size_t)(row0 + l15) * K + 16 * kgrp;
    const _Float16* bcol = Bw + (size_t)(col0 + l15) * K + 16 * kgrp;

    v8f acc = {};
    for (int kk = 0; kk < K; kk += 32) {
        v16h a = load_fragA(arow, kk);
        v16h b = *(const v16h*)(bcol + kk);
        acc = __builtin_amdgcn_wmma_f32_16x16x32_f16(
            false, a, false, b, (short)0, acc, false, false);
    }
    // D layout: n = lane&15, m = v + 8*(lane>>4)
    const int n = col0 + l15;
    const float bn = bias[n];
#pragma unroll
    for (int v = 0; v < 8; ++v) {
        int m = row0 + v + 8 * kgrp;
        C[(size_t)m * N + n] = acc[v] + bn;
    }
}

// ---------------------------------------------------------------------------
// LIF on the input layer (elementwise over MSTEP x H0), emits f16 spikes that
// feed the hid WMMA GEMM as A-operand.
// ---------------------------------------------------------------------------
__global__ void __launch_bounds__(256)
k_lif_in(const float* __restrict__ cur, float* __restrict__ mem,
         const float* __restrict__ beta, const float* __restrict__ thr,
         _Float16* __restrict__ spk, int total) {
    int idx = blockIdx.x * 256 + threadIdx.x;
    if (idx >= total) return;
    int h = idx & (H0 - 1);
    float th = thr[h];
    float bb = clamp01f(beta[h]);
    float mm = mem[idx];
    float reset = (mm > th) ? 1.f : 0.f;              // uses previous mem
    mm = bb * mm + cur[idx] - reset * th;
    mem[idx] = mm;
    spk[idx] = (_Float16)(((mm - th) > 0.f) ? 1.f : 0.f);
}

// ---------------------------------------------------------------------------
// hid GEMM + fused LIF epilogue:
//   cur = spk_in[M,K] * W_hid[N,K]^T + b_hid ; LIF(mem_hid) ; emit spk f16
// ---------------------------------------------------------------------------
__global__ void __launch_bounds__(256)
k_gemm_hid_lif(const _Float16* __restrict__ A, const _Float16* __restrict__ Bw,
               const float* __restrict__ bias, const float* __restrict__ beta,
               const float* __restrict__ thr, float* __restrict__ mem,
               _Float16* __restrict__ spk, int M, int N, int K) {
    const int lane = threadIdx.x & 31;
    const int wave = threadIdx.x >> 5;
    const int tile = blockIdx.x * 8 + wave;
    const int ntn  = N >> 4;
    const int row0 = (tile / ntn) << 4;
    const int col0 = (tile % ntn) << 4;
    const int kgrp = lane >> 4;
    const int l15  = lane & 15;

    const _Float16* arow = A  + (size_t)(row0 + l15) * K + 16 * kgrp;
    const _Float16* bcol = Bw + (size_t)(col0 + l15) * K + 16 * kgrp;

    v8f acc = {};
    for (int kk = 0; kk < K; kk += 32) {
        v16h a = load_fragA(arow, kk);
        v16h b = *(const v16h*)(bcol + kk);
        acc = __builtin_amdgcn_wmma_f32_16x16x32_f16(
            false, a, false, b, (short)0, acc, false, false);
    }
    const int n = col0 + l15;
    const float bn = bias[n];
    const float th = thr[n];
    const float bb = clamp01f(beta[n]);
#pragma unroll
    for (int v = 0; v < 8; ++v) {
        int m = row0 + v + 8 * kgrp;
        size_t idx = (size_t)m * N + n;
        float c = acc[v] + bn;
        float mm = mem[idx];
        float reset = (mm > th) ? 1.f : 0.f;
        mm = bb * mm + c - reset * th;
        mem[idx] = mm;
        spk[idx] = (_Float16)(((mm - th) > 0.f) ? 1.f : 0.f);
    }
}

// ---------------------------------------------------------------------------
// Output layer: cur_out[b,c] = spk_hid[b, 0:5376] . W_out[c, :] + b_out[c],
// fused LIF (thr = 1.0). N=6 is too narrow for WMMA; block-per-row reduce.
// ---------------------------------------------------------------------------
__global__ void __launch_bounds__(256)
k_out_lif(const _Float16* __restrict__ spkH, const _Float16* __restrict__ Wout,
          const float* __restrict__ bias, const float* __restrict__ beta,
          float* __restrict__ mem, float* __restrict__ outp) {
    __shared__ float red[NCLS][256];
    const int b = blockIdx.x, tid = threadIdx.x;
    const _Float16* srow = spkH + (size_t)b * FLATK;
    float acc[NCLS] = {};
    for (int k = tid; k < FLATK; k += 256) {
        float s = (float)srow[k];                 // exactly 0.0 or 1.0
#pragma unroll
        for (int c = 0; c < NCLS; ++c)
            acc[c] += s * (float)Wout[c * FLATK + k];
    }
#pragma unroll
    for (int c = 0; c < NCLS; ++c) red[c][tid] = acc[c];
    __syncthreads();
    for (int off = 128; off > 0; off >>= 1) {
        if (tid < off) {
#pragma unroll
            for (int c = 0; c < NCLS; ++c) red[c][tid] += red[c][tid + off];
        }
        __syncthreads();
    }
    if (tid < NCLS) {
        float cur = red[tid][0] + bias[tid];
        float mm  = mem[b * NCLS + tid];
        float reset = (mm > 1.0f) ? 1.f : 0.f;
        mm = clamp01f(beta[tid]) * mm + cur - reset * 1.0f;
        mem[b * NCLS + tid] = mm;
        outp[b * NCLS + tid] = ((mm - 1.0f) > 0.f) ? 1.f : 0.f;
    }
}

// ---------------------------------------------------------------------------
static inline size_t alignup256(size_t v) { return (v + 255) & ~(size_t)255; }

extern "C" void kernel_launch(void* const* d_in, const int* in_sizes, int n_in,
                              void* d_out, int out_size, void* d_ws, size_t ws_size,
                              hipStream_t stream) {
    const float* x        = (const float*)d_in[0];
    const float* W_in     = (const float*)d_in[1];
    const float* b_in     = (const float*)d_in[2];
    const float* W_hid    = (const float*)d_in[3];
    const float* b_hid    = (const float*)d_in[4];
    const float* W_out    = (const float*)d_in[5];
    const float* b_out    = (const float*)d_in[6];
    const float* beta_in  = (const float*)d_in[7];
    const float* thr_in   = (const float*)d_in[8];
    const float* beta_hid = (const float*)d_in[9];
    const float* thr_hid  = (const float*)d_in[10];
    const float* beta_out = (const float*)d_in[11];
    float* out = (float*)d_out;

    // ---- workspace carve-up (~40 MB, all 256B aligned) ----
    char* ws = (char*)d_ws;
    size_t off = 0;
    _Float16* prA = (_Float16*)(ws + off); off = alignup256(off + (size_t)MFULL * KIN * 2);
    float* curin  = (float*)   (ws + off); off = alignup256(off + (size_t)MFULL * H0 * 4);
    _Float16* Win16  = (_Float16*)(ws + off); off = alignup256(off + (size_t)H0 * KIN * 2);
    _Float16* Whid16 = (_Float16*)(ws + off); off = alignup256(off + (size_t)H1 * H0 * 2);
    _Float16* Wout16 = (_Float16*)(ws + off); off = alignup256(off + (size_t)NCLS * FLATK * 2);
    float* mem_in  = (float*)(ws + off); off = alignup256(off + (size_t)MSTEP * H0 * 4);
    float* mem_hid = (float*)(ws + off); off = alignup256(off + (size_t)MSTEP * H1 * 4);
    float* mem_out = (float*)(ws + off); off = alignup256(off + (size_t)BATCH * NCLS * 4);
    _Float16* spkA = (_Float16*)(ws + off); off = alignup256(off + (size_t)MSTEP * H0 * 2);
    _Float16* spkH = (_Float16*)(ws + off); off = alignup256(off + (size_t)MSTEP * H1 * 2);
    (void)ws_size; (void)in_sizes; (void)n_in; (void)out_size;

    // ---- one-time-per-call prep: weight f16 copies + zeroed membrane state ----
    k_f32_to_f16<<<(H0 * KIN + 255) / 256, 256, 0, stream>>>(W_in,  Win16,  H0 * KIN);
    k_f32_to_f16<<<(H1 * H0  + 255) / 256, 256, 0, stream>>>(W_hid, Whid16, H1 * H0);
    k_f32_to_f16<<<(NCLS * FLATK + 255) / 256, 256, 0, stream>>>(W_out, Wout16, NCLS * FLATK);
    k_zero_f32<<<(MSTEP * H0 + 255) / 256, 256, 0, stream>>>(mem_in,  MSTEP * H0);
    k_zero_f32<<<(MSTEP * H1 + 255) / 256, 256, 0, stream>>>(mem_hid, MSTEP * H1);
    k_zero_f32<<<(BATCH * NCLS + 255) / 256, 256, 0, stream>>>(mem_out, BATCH * NCLS);

    // ---- pool: one streaming pass over x (the bandwidth-dominant step) ----
    const int poolTotal = MFULL * KIN;          // 5,505,024
    k_pool<<<poolTotal / 256, 256, 0, stream>>>(x, prA, poolTotal);

    // ---- fc_in for all T at once: (21504x256)x(256x256), WMMA f16 ----
    // tiles = (21504/16)*(256/16) = 21504; 8 waves/block -> 2688 blocks exact
    k_gemm_fcin<<<(MFULL / 16) * (H0 / 16) / 8, 256, 0, stream>>>(
        prA, Win16, b_in, curin, MFULL, H0, KIN);

    // ---- sequential timestep chain ----
    for (int t = 0; t < T_STEPS; ++t) {
        const float* cur_t = curin + (size_t)t * MSTEP * H0;
        k_lif_in<<<(MSTEP * H0) / 256, 256, 0, stream>>>(
            cur_t, mem_in, beta_in, thr_in, spkA, MSTEP * H0);
        // tiles = (2688/16)*(128/16) = 1344 -> 168 blocks exact
        k_gemm_hid_lif<<<(MSTEP / 16) * (H1 / 16) / 8, 256, 0, stream>>>(
            spkA, Whid16, b_hid, beta_hid, thr_hid, mem_hid, spkH,
            MSTEP, H1, H0);
        k_out_lif<<<BATCH, 256, 0, stream>>>(
            spkH, Wout16, b_out, beta_out, mem_out,
            out + (size_t)t * BATCH * NCLS);
    }
}
